// ChannelAttention_42331197669736
// MI455X (gfx1250) — matrix operand order repaired and checked by reading, not compiled, not run on hardware
//
#include <hip/hip_runtime.h>
#include <hip/hip_bf16.h>

typedef __attribute__((ext_vector_type(16))) _Float16 v16h;
typedef __attribute__((ext_vector_type(8)))  _Float16 v8h;
typedef __attribute__((ext_vector_type(4)))  _Float16 v4h;
typedef __attribute__((ext_vector_type(2)))  _Float16 v2h;
typedef __attribute__((ext_vector_type(8)))  float    v8f;
typedef __attribute__((ext_vector_type(4)))  float    v4f;

// Problem dims (fixed by the reference)
constexpr int NB = 16;    // batch
constexpr int CC = 512;   // channels
constexpr int LL = 4096;  // H*W

// Block tiling: 128(M) x 128(N) per block, K-step 32.
// 256 threads = 8 waves arranged 4(M) x 2(N); each wave owns a 32x64 patch = 2x4 WMMA tiles.
// Per K-step per wave: 2 A-frags + 4 B-frags (6x 2x ds_load_b128) feed 8 WMMAs.
constexpr int BM = 128;
constexpr int BN = 128;
constexpr int BK = 32;
constexpr int SA = BK + 8;     // LDS row stride for A tile (halfs); 80B rows, 16B-aligned chunks
constexpr int BSTR = 24;       // B fragment lane stride (halfs) = 48B, 16B-aligned, padded
constexpr int BS_ELEMS = (BN / 16) * 32 * BSTR;   // 8 groups x 32 lanes x 24 halfs

__device__ __attribute__((always_inline)) inline v8f wmma_f16(v16h a, v16h b, v8f c) {
  // (neg_a, A, neg_b, B, c_mod, C, reuse_a, reuse_b)
  return __builtin_amdgcn_wmma_f32_16x16x32_f16(false, a, false, b, (short)0, c, false, false);
}

// Swizzled (fragment-major) LDS index for B tile cell (k, c), k in [0,32), c in [0,128).
// Lane (hs*16+n) reading column group g finds its 16 K-halfs contiguous.
__device__ __attribute__((always_inline)) inline int bs_idx(int k, int c) {
  return (((c >> 4) << 5) + ((k >> 4) << 4) + (c & 15)) * BSTR + (k & 15);
}

// A fragment: 16x32 (MxK) f16 from row-major LDS tile. Two 16B contiguous chunks.
__device__ __attribute__((always_inline)) inline v16h load_frag_a(const _Float16* As, int m_base) {
  const int lane = threadIdx.x & 31;
  const int hs   = lane >> 4;
  const int m    = m_base + (lane & 15);
  const v8h lo = *(const v8h*)&As[m * SA + hs * 16];
  const v8h hi = *(const v8h*)&As[m * SA + hs * 16 + 8];
  return __builtin_shufflevector(lo, hi, 0, 1, 2, 3, 4, 5, 6, 7, 8, 9, 10, 11, 12, 13, 14, 15);
}

// B fragment: 32x16 (KxN) f16 from swizzled LDS tile: one 32B contiguous run per lane.
__device__ __attribute__((always_inline)) inline v16h load_frag_b(const _Float16* Bs, int n_base) {
  const int lane = threadIdx.x & 31;
  const _Float16* p = Bs + (size_t)(((n_base >> 4) << 5) + lane) * BSTR;
  const v8h lo = *(const v8h*)(p);
  const v8h hi = *(const v8h*)(p + 8);
  return __builtin_shufflevector(lo, hi, 0, 1, 2, 3, 4, 5, 6, 7, 8, 9, 10, 11, 12, 13, 14, 15);
}

// ---------------------------------------------------------------------------
// Kernel 1: QKV projections.  P = W * X  per (batch, weight-select).
//   wsel==0 -> Q  [CC,LL] f16
//   wsel==1 -> K^T[LL,CC] f16   (transposed so scores GEMM is row-major x row-major)
//   wsel==2 -> V  [CC,LL] f16
// ---------------------------------------------------------------------------
__global__ void __launch_bounds__(256)
qkv_gemm(const float* __restrict__ x,
         const float* __restrict__ Wq, const float* __restrict__ Wk, const float* __restrict__ Wv,
         _Float16* __restrict__ Qh, _Float16* __restrict__ Kth, _Float16* __restrict__ Vh) {
  const int z     = blockIdx.z;
  const int wsel  = z % 3;
  const int batch = z / 3;
  const float* W = (wsel == 0) ? Wq : ((wsel == 1) ? Wk : Wv);
  const float* X = x + (size_t)batch * CC * LL;

  const int m0 = blockIdx.y * BM;
  const int n0 = blockIdx.x * BN;

  __shared__ alignas(16) _Float16 As[2][BM * SA];
  __shared__ alignas(16) _Float16 Bs[2][BS_ELEMS];

  const int tid  = threadIdx.x;
  const int wave = tid >> 5;
  const int wm   = wave >> 1;   // 0..3
  const int wn   = wave & 1;    // 0..1

  v8f acc[2][4] = {};

  // Double-buffered tile staging: all global loads first (one wait), then cvt+store.
  auto stage = [&](int k0, int buf) {
    // A tile: 128x32 weights, 4 float4 chunks per thread
    v4f wreg[4];
#pragma unroll
    for (int it = 0; it < 4; ++it) {
      const int chunk = it * 256 + tid;      // 0..1023
      const int r  = chunk >> 3;             // 0..127
      const int c4 = (chunk & 7) * 4;        // 0..28
      wreg[it] = *(const v4f*)&W[(size_t)(m0 + r) * CC + (k0 + c4)];
    }
    // B tile: 32x128 activations, per thread 2 adjacent K rows x 4 cols, twice
    v4f xr0[2], xr1[2];
#pragma unroll
    for (int it = 0; it < 2; ++it) {
      const int chunk = it * 256 + tid;      // 0..511
      const int kp = ((chunk >> 5) & 15) * 2;  // 0,2,..,30
      const int c4 = (chunk & 31) * 4;         // 0..124
      xr0[it] = *(const v4f*)&X[(size_t)(k0 + kp)     * LL + (n0 + c4)];
      xr1[it] = *(const v4f*)&X[(size_t)(k0 + kp + 1) * LL + (n0 + c4)];
    }
    if (k0 + BK < CC)  // hint the next tile -> global_prefetch_b8
      __builtin_prefetch(&X[(size_t)(k0 + BK + (tid >> 3)) * LL + n0 + ((tid & 7) << 4)], 0, 1);
    // convert + store phase
#pragma unroll
    for (int it = 0; it < 4; ++it) {
      const int chunk = it * 256 + tid;
      const int r  = chunk >> 3;
      const int c4 = (chunk & 7) * 4;
      v4h h;
#pragma unroll
      for (int j = 0; j < 4; ++j) h[j] = (_Float16)wreg[it][j];
      *(v4h*)&As[buf][r * SA + c4] = h;
    }
#pragma unroll
    for (int it = 0; it < 2; ++it) {
      const int chunk = it * 256 + tid;
      const int kp = ((chunk >> 5) & 15) * 2;
      const int c4 = (chunk & 31) * 4;
#pragma unroll
      for (int j = 0; j < 4; ++j) {
        v2h p;                                  // packed (k, k+1) pair -> ds_store_b32
        p[0] = (_Float16)xr0[it][j];
        p[1] = (_Float16)xr1[it][j];
        *(v2h*)&Bs[buf][bs_idx(kp, c4 + j)] = p;
      }
    }
  };

  stage(0, 0);
  __syncthreads();
  constexpr int NT = CC / BK;   // 16 K-tiles
  for (int t = 0; t < NT; ++t) {
    const int buf = t & 1;
    if (t + 1 < NT) stage((t + 1) * BK, (t + 1) & 1);

    const v16h a0 = load_frag_a(&As[buf][0], wm * 32);
    const v16h a1 = load_frag_a(&As[buf][0], wm * 32 + 16);
#pragma unroll
    for (int tn = 0; tn < 4; ++tn) {
      const v16h b = load_frag_b(&Bs[buf][0], wn * 64 + tn * 16);
      acc[0][tn] = wmma_f16(a0, b, acc[0][tn]);
      acc[1][tn] = wmma_f16(a1, b, acc[1][tn]);
    }
    __syncthreads();
  }

  // store 16x16 f32 accumulators as f16 per the C/D layout
  const int lane = tid & 31;
  const int hs   = lane >> 4;
  const int nn   = lane & 15;
#pragma unroll
  for (int tm = 0; tm < 2; ++tm)
#pragma unroll
    for (int tn = 0; tn < 4; ++tn)
#pragma unroll
      for (int r = 0; r < 8; ++r) {
        const int gm = m0 + wm * 32 + tm * 16 + r + 8 * hs;
        const int gn = n0 + wn * 64 + tn * 16 + nn;
        const _Float16 v = (_Float16)acc[tm][tn][r];
        if (wsel == 0)      Qh [((size_t)batch * CC + gm) * LL + gn] = v;
        else if (wsel == 1) Kth[((size_t)batch * LL + gn) * CC + gm] = v;   // transposed
        else                Vh [((size_t)batch * CC + gm) * LL + gn] = v;
      }
}

// ---------------------------------------------------------------------------
// Kernel 2: scores = Q * K^T.  A=[CC,LL] f16, B=[LL,CC] f16, D=[CC,CC] f32.
// ---------------------------------------------------------------------------
__global__ void __launch_bounds__(256)
scores_gemm(const _Float16* __restrict__ Qh, const _Float16* __restrict__ Kth,
            float* __restrict__ S) {
  const int batch = blockIdx.z;
  const _Float16* A = Qh  + (size_t)batch * CC * LL;
  const _Float16* B = Kth + (size_t)batch * LL * CC;

  const int m0 = blockIdx.y * BM;
  const int n0 = blockIdx.x * BN;

  __shared__ alignas(16) _Float16 As[2][BM * SA];
  __shared__ alignas(16) _Float16 Bs[2][BS_ELEMS];

  const int tid  = threadIdx.x;
  const int wave = tid >> 5;
  const int wm   = wave >> 1;
  const int wn   = wave & 1;

  v8f acc[2][4] = {};

  auto stage = [&](int k0, int buf) {
    // A: 128x32 halfs, two b128 load+store per thread
    v8h q[2];
#pragma unroll
    for (int it = 0; it < 2; ++it) {
      const int chunk = it * 256 + tid;      // 0..511
      const int ra  = chunk >> 2;            // 0..127
      const int ca8 = (chunk & 3) * 8;       // 0..24
      q[it] = *(const v8h*)&A[(size_t)(m0 + ra) * LL + (k0 + ca8)];
    }
    // B: 2 adjacent K rows x 8 cols per thread (4096 halfs total)
    const int kp  = (tid >> 4) * 2;          // 0..30
    const int cb8 = (tid & 15) * 8;          // 0..120
    const v8h b0 = *(const v8h*)&B[(size_t)(k0 + kp)     * CC + (n0 + cb8)];
    const v8h b1 = *(const v8h*)&B[(size_t)(k0 + kp + 1) * CC + (n0 + cb8)];
#pragma unroll
    for (int it = 0; it < 2; ++it) {
      const int chunk = it * 256 + tid;
      const int ra  = chunk >> 2;
      const int ca8 = (chunk & 3) * 8;
      *(v8h*)&As[buf][ra * SA + ca8] = q[it];
    }
#pragma unroll
    for (int j = 0; j < 8; ++j) {
      v2h p; p[0] = b0[j]; p[1] = b1[j];     // ds_store_b32
      *(v2h*)&Bs[buf][bs_idx(kp, cb8 + j)] = p;
    }
  };

  stage(0, 0);
  __syncthreads();
  constexpr int NT = LL / BK;   // 128 K-tiles
  for (int t = 0; t < NT; ++t) {
    const int buf = t & 1;
    if (t + 1 < NT) stage((t + 1) * BK, (t + 1) & 1);

    const v16h a0 = load_frag_a(&As[buf][0], wm * 32);
    const v16h a1 = load_frag_a(&As[buf][0], wm * 32 + 16);
#pragma unroll
    for (int tn = 0; tn < 4; ++tn) {
      const v16h b = load_frag_b(&Bs[buf][0], wn * 64 + tn * 16);
      acc[0][tn] = wmma_f16(a0, b, acc[0][tn]);
      acc[1][tn] = wmma_f16(a1, b, acc[1][tn]);
    }
    __syncthreads();
  }

  const int lane = tid & 31;
  const int hs   = lane >> 4;
  const int nn   = lane & 15;
#pragma unroll
  for (int tm = 0; tm < 2; ++tm)
#pragma unroll
    for (int tn = 0; tn < 4; ++tn)
#pragma unroll
      for (int r = 0; r < 8; ++r) {
        const int gm = m0 + wm * 32 + tm * 16 + r + 8 * hs;
        const int gn = n0 + wn * 64 + tn * 16 + nn;
        S[((size_t)batch * CC + gm) * CC + gn] = acc[tm][tn][r];
      }
}

// ---------------------------------------------------------------------------
// Kernel 3: row softmax over scores [NB*CC rows of CC], output f16 probs.
// ---------------------------------------------------------------------------
__global__ void __launch_bounds__(256)
softmax_rows(const float* __restrict__ S, _Float16* __restrict__ Ah) {
  const size_t row = blockIdx.x;
  const float* s = S + row * CC;
  __shared__ float red[256];
  const int t = threadIdx.x;

  const float v0 = s[t];
  const float v1 = s[t + 256];

  red[t] = fmaxf(v0, v1);
  __syncthreads();
#pragma unroll
  for (int off = 128; off > 0; off >>= 1) {
    if (t < off) red[t] = fmaxf(red[t], red[t + off]);
    __syncthreads();
  }
  const float rmax = red[0];
  __syncthreads();

  const float e0 = __expf(v0 - rmax);
  const float e1 = __expf(v1 - rmax);
  red[t] = e0 + e1;
  __syncthreads();
#pragma unroll
  for (int off = 128; off > 0; off >>= 1) {
    if (t < off) red[t] += red[t + off];
    __syncthreads();
  }
  const float inv = 1.0f / red[0];

  _Float16* o = Ah + row * CC;
  o[t]       = (_Float16)(e0 * inv);
  o[t + 256] = (_Float16)(e1 * inv);
}

// ---------------------------------------------------------------------------
// Kernel 4: out = gama * (A_soft * V) + x.  A=[CC,CC] f16, B=[CC,LL] f16.
// ---------------------------------------------------------------------------
__global__ void __launch_bounds__(256)
out_gemm(const _Float16* __restrict__ Ah, const _Float16* __restrict__ Vh,
         const float* __restrict__ x, const float* __restrict__ gama,
         float* __restrict__ out) {
  const int batch = blockIdx.z;
  const _Float16* A = Ah + (size_t)batch * CC * CC;
  const _Float16* B = Vh + (size_t)batch * CC * LL;

  const int m0 = blockIdx.y * BM;
  const int n0 = blockIdx.x * BN;

  __shared__ alignas(16) _Float16 As[2][BM * SA];
  __shared__ alignas(16) _Float16 Bs[2][BS_ELEMS];

  const int tid  = threadIdx.x;
  const int wave = tid >> 5;
  const int wm   = wave >> 1;
  const int wn   = wave & 1;

  v8f acc[2][4] = {};

  auto stage = [&](int k0, int buf) {
    v8h q[2];
#pragma unroll
    for (int it = 0; it < 2; ++it) {
      const int chunk = it * 256 + tid;
      const int ra  = chunk >> 2;
      const int ca8 = (chunk & 3) * 8;
      q[it] = *(const v8h*)&A[(size_t)(m0 + ra) * CC + (k0 + ca8)];
    }
    const int kp  = (tid >> 4) * 2;
    const int cb8 = (tid & 15) * 8;
    const v8h b0 = *(const v8h*)&B[(size_t)(k0 + kp)     * LL + (n0 + cb8)];
    const v8h b1 = *(const v8h*)&B[(size_t)(k0 + kp + 1) * LL + (n0 + cb8)];
#pragma unroll
    for (int it = 0; it < 2; ++it) {
      const int chunk = it * 256 + tid;
      const int ra  = chunk >> 2;
      const int ca8 = (chunk & 3) * 8;
      *(v8h*)&As[buf][ra * SA + ca8] = q[it];
    }
#pragma unroll
    for (int j = 0; j < 8; ++j) {
      v2h p; p[0] = b0[j]; p[1] = b1[j];
      *(v2h*)&Bs[buf][bs_idx(kp, cb8 + j)] = p;
    }
  };

  stage(0, 0);
  __syncthreads();
  constexpr int NT = CC / BK;   // 16 K-tiles
  for (int t = 0; t < NT; ++t) {
    const int buf = t & 1;
    if (t + 1 < NT) stage((t + 1) * BK, (t + 1) & 1);

    const v16h a0 = load_frag_a(&As[buf][0], wm * 32);
    const v16h a1 = load_frag_a(&As[buf][0], wm * 32 + 16);
#pragma unroll
    for (int tn = 0; tn < 4; ++tn) {
      const v16h b = load_frag_b(&Bs[buf][0], wn * 64 + tn * 16);
      acc[0][tn] = wmma_f16(a0, b, acc[0][tn]);
      acc[1][tn] = wmma_f16(a1, b, acc[1][tn]);
    }
    __syncthreads();
  }

  const float g    = gama[0];
  const int   lane = tid & 31;
  const int   hs   = lane >> 4;
  const int   nn   = lane & 15;
#pragma unroll
  for (int tm = 0; tm < 2; ++tm)
#pragma unroll
    for (int tn = 0; tn < 4; ++tn)
#pragma unroll
      for (int r = 0; r < 8; ++r) {
        const int gm = m0 + wm * 32 + tm * 16 + r + 8 * hs;
        const int gn = n0 + wn * 64 + tn * 16 + nn;
        const size_t idx = ((size_t)batch * CC + gm) * LL + gn;
        out[idx] = g * acc[tm][tn][r] + x[idx];
      }
}

// ---------------------------------------------------------------------------
extern "C" void kernel_launch(void* const* d_in, const int* in_sizes, int n_in,
                              void* d_out, int out_size, void* d_ws, size_t ws_size,
                              hipStream_t stream) {
  const float* x    = (const float*)d_in[0];
  const float* Wq   = (const float*)d_in[1];
  const float* Wk   = (const float*)d_in[2];
  const float* Wv   = (const float*)d_in[3];
  const float* gama = (const float*)d_in[4];
  float*       out  = (float*)d_out;

  // Workspace carve-out (bytes)
  char* ws = (char*)d_ws;
  const size_t qkv_elems = (size_t)NB * CC * LL;         // 33,554,432
  _Float16* Qh  = (_Float16*)(ws);                       //  64 MB
  _Float16* Kth = (_Float16*)(ws + qkv_elems * 2);       //  64 MB
  _Float16* Vh  = (_Float16*)(ws + qkv_elems * 4);       //  64 MB
  float*    S   = (float*)   (ws + qkv_elems * 6);       //  16 MB  [NB,CC,CC] f32
  _Float16* Ah  = (_Float16*)(ws + qkv_elems * 6 + (size_t)NB * CC * CC * 4); // 8 MB

  // 1) Q, K^T, V projections (f32 -> f16 staged through LDS, WMMA f16)
  qkv_gemm<<<dim3(LL / BN, CC / BM, NB * 3), 256, 0, stream>>>(x, Wq, Wk, Wv, Qh, Kth, Vh);
  // 2) scores = Q * K^T  (K-dim = 4096)
  scores_gemm<<<dim3(CC / BN, CC / BM, NB), 256, 0, stream>>>(Qh, Kth, S);
  // 3) softmax over channel dim, emit f16 probabilities
  softmax_rows<<<dim3(NB * CC), 256, 0, stream>>>(S, Ah);
  // 4) out = gama * (A * V) + x
  out_gemm<<<dim3(LL / BN, CC / BM, NB), 256, 0, stream>>>(Ah, Vh, x, gama, out);
}
